// LambdaLayer3d_73607149519615
// MI455X (gfx1250) — matrix operand order repaired and checked by reading, not compiled, not run on hardware
//
#include <hip/hip_runtime.h>
#include <math.h>

// ---------------- problem constants ----------------
#define C_IN   128
#define DDIM   48
#define NPOS   (48*48*48)      // 110592
#define OTOT   112             // stacked proj rows: q[0..63], k[64..79], v[80..111]
#define K_OFF  64
#define V_OFF  80
#define KD     16
#define HEADS  4
#define VD     32
#define EPS_BN 1e-5f

typedef __bf16 v16bf __attribute__((ext_vector_type(16)));
typedef float  v8f   __attribute__((ext_vector_type(8)));

static __device__ __forceinline__ unsigned short f2bf(float f) {
  union { float f; unsigned u; } c; c.f = f;
  unsigned r = c.u + 0x7FFFu + ((c.u >> 16) & 1u);   // RNE
  return (unsigned short)(r >> 16);
}
static __device__ __forceinline__ __bf16 us2bf(unsigned short u) {
  union { unsigned short u; __bf16 b; } c; c.u = u; return c.b;
}

// ---------------- pack W (Wq|Wk|Wv) to bf16 ----------------
__global__ void convert_w_kernel(const float* __restrict__ Wq,
                                 const float* __restrict__ Wk,
                                 const float* __restrict__ Wv,
                                 unsigned short* __restrict__ wbf) {
  int i = blockIdx.x * blockDim.x + threadIdx.x;
  if (i >= OTOT * C_IN) return;
  int o = i / C_IN, c = i % C_IN;
  float v;
  if (o < 64)      v = Wq[o * C_IN + c];
  else if (o < 80) v = Wk[(o - 64) * C_IN + c];
  else             v = Wv[(o - 80) * C_IN + c];
  wbf[i] = f2bf(v);
}

// ---------------- x fp32 -> bf16 ----------------
__global__ void convert_x_kernel(const float* __restrict__ x,
                                 unsigned short* __restrict__ xbf,
                                 long long total) {
  long long i = (long long)blockIdx.x * blockDim.x + threadIdx.x;
  if (i < total) xbf[i] = f2bf(x[i]);
}

// ---------------- projection GEMM via WMMA bf16 ----------------
// grid (NPOS/16, b), block 32 (one wave). Each wave: 16-position tile,
// 7 output row-tiles, K-loop 128/32 -> 28 x v_wmma_f32_16x16x32_bf16.
__global__ void proj_wmma_kernel(const unsigned short* __restrict__ xbf,
                                 const unsigned short* __restrict__ wbf,
                                 float* __restrict__ proj) {
  const int tile = blockIdx.x;
  const int bidx = blockIdx.y;
  const int lane = threadIdx.x;          // 0..31
  const int pos0 = tile * 16;
  const unsigned short* xb = xbf + (size_t)bidx * C_IN * NPOS;

  v8f acc[7];
  #pragma unroll
  for (int t = 0; t < 7; ++t) acc[t] = (v8f){0.f,0.f,0.f,0.f,0.f,0.f,0.f,0.f};

  const int mrow = lane & 15;
  const int hiA  = (lane < 16) ? 0 : 8;    // A: K-group offset per half-wave
  const int hiB  = (lane < 16) ? 0 : 16;   // unused; B load also uses hiA
  const int ncol = lane & 15;              // B/C/D column

  for (int kk = 0; kk < C_IN; kk += 32) {
    // B tile: x[kk..kk+31][pos0..pos0+15] (ISA B layout, K=32 bf16)
    v16bf bm;
    const unsigned short* xcol = xb + pos0 + ncol;
    #pragma unroll
    for (int i = 0; i < 16; ++i)
      bm[i] = us2bf(xcol[(size_t)(kk + hiA + i + ((i < 8) ? 0 : 8)) * NPOS]);

    #pragma unroll
    for (int t = 0; t < 7; ++t) {
      // A tile: W[t*16..+15][kk..kk+31] (ISA 16-bit A layout)
      v16bf am;
      const unsigned short* wr = wbf + (t * 16 + mrow) * C_IN + kk;
      #pragma unroll
      for (int i = 0; i < 16; ++i) {
        int k = i + ((i < 8) ? 0 : 8) + hiA;
        am[i] = us2bf(wr[k]);
      }
      acc[t] = __builtin_amdgcn_wmma_f32_16x16x32_bf16(
          false, am, false, bm, (short)0, acc[t], false, false);
    }
  }

  const int rowadd = (lane < 16) ? 0 : 8;
  #pragma unroll
  for (int t = 0; t < 7; ++t) {
    #pragma unroll
    for (int r = 0; r < 8; ++r) {
      int orow = t * 16 + r + rowadd;
      proj[((size_t)bidx * OTOT + orow) * NPOS + pos0 + ncol] = acc[t][r];
    }
  }
}

// ---------------- BN stats: 64 q channels + 32 v channels ----------------
__global__ void bn_stats_kernel(const float* __restrict__ proj,
                                float* __restrict__ stats /* [96][2] */) {
  __shared__ float ssum[256], ssq[256];
  const int s = blockIdx.x;                       // 0..95
  const int row = (s < 64) ? s : (V_OFF + (s - 64));
  const int tid = threadIdx.x;
  float sum = 0.f, sq = 0.f;
  for (int b = 0; b < 2; ++b) {
    const float* p = proj + ((size_t)b * OTOT + row) * NPOS;
    for (int i = tid; i < NPOS; i += 256) { float v = p[i]; sum += v; sq += v * v; }
  }
  ssum[tid] = sum; ssq[tid] = sq;
  __syncthreads();
  for (int st = 128; st > 0; st >>= 1) {
    if (tid < st) { ssum[tid] += ssum[tid + st]; ssq[tid] += ssq[tid + st]; }
    __syncthreads();
  }
  if (tid == 0) {
    float inv = 1.f / (2.f * (float)NPOS);
    float m = ssum[0] * inv;
    stats[2 * s]     = m;
    stats[2 * s + 1] = ssq[0] * inv - m * m;
  }
}

// ---------------- softmax stats for 16 k rows per batch ----------------
__global__ void softmax_stats_kernel(const float* __restrict__ proj,
                                     float* __restrict__ sstats /* [32][2] */) {
  __shared__ float red[256];
  const int id = blockIdx.x;                 // 0..31: b = id>>4, k = id&15
  const int b = id >> 4, kk = id & 15;
  const int tid = threadIdx.x;
  const float* p = proj + ((size_t)b * OTOT + K_OFF + kk) * NPOS;
  float mx = -3.0e38f;
  for (int i = tid; i < NPOS; i += 256) mx = fmaxf(mx, p[i]);
  red[tid] = mx; __syncthreads();
  for (int st = 128; st > 0; st >>= 1) {
    if (tid < st) red[tid] = fmaxf(red[tid], red[tid + st]);
    __syncthreads();
  }
  const float gmx = red[0];
  __syncthreads();
  float s = 0.f;
  for (int i = tid; i < NPOS; i += 256) s += expf(p[i] - gmx);
  red[tid] = s; __syncthreads();
  for (int st = 128; st > 0; st >>= 1) {
    if (tid < st) red[tid] += red[tid + st];
    __syncthreads();
  }
  if (tid == 0) { sstats[2 * id] = gmx; sstats[2 * id + 1] = red[0]; }
}

// ---------------- lam_c[b,16,32] = softmax(k) @ bn(v)^T ----------------
#define LC_CH   64
#define LC_POSB 4096
__global__ void lam_c_kernel(const float* __restrict__ proj,
                             const float* __restrict__ stats,
                             const float* __restrict__ sstats,
                             const float* __restrict__ gv, const float* __restrict__ bv,
                             float* __restrict__ lamc) {
  __shared__ float skt[KD * LC_CH];
  __shared__ float svt[VD * LC_CH];
  __shared__ float smx[KD], srs[KD], vsc[VD], vsh[VD];
  const int b = blockIdx.y;
  const int tid = threadIdx.x;

  if (tid < KD) {
    smx[tid] = sstats[2 * (b * KD + tid)];
    srs[tid] = 1.f / sstats[2 * (b * KD + tid) + 1];
  } else if (tid < KD + VD) {
    int v = tid - KD;
    float m = stats[2 * (64 + v)], va = stats[2 * (64 + v) + 1];
    float sc = gv[v] * rsqrtf(va + EPS_BN);
    vsc[v] = sc; vsh[v] = bv[v] - m * sc;
  }

  const int k0 = tid >> 5,        v0 = tid & 31;
  const int k1 = (tid + 256) >> 5, v1 = tid & 31;
  float a0 = 0.f, a1 = 0.f;
  const float* kb = proj + ((size_t)b * OTOT + K_OFF) * NPOS;
  const float* vb = proj + ((size_t)b * OTOT + V_OFF) * NPOS;
  const int base0 = blockIdx.x * LC_POSB;

  for (int c = 0; c < LC_POSB / LC_CH; ++c) {
    const int base = base0 + c * LC_CH;
    __syncthreads();
    for (int e = tid; e < KD * LC_CH; e += 256) {
      int kk = e / LC_CH, p = e % LC_CH;
      skt[e] = expf(kb[(size_t)kk * NPOS + base + p] - smx[kk]) * srs[kk];
    }
    for (int e = tid; e < VD * LC_CH; e += 256) {
      int vv = e / LC_CH, p = e % LC_CH;
      svt[e] = vb[(size_t)vv * NPOS + base + p] * vsc[vv] + vsh[vv];
    }
    __syncthreads();
    #pragma unroll 4
    for (int p = 0; p < LC_CH; ++p) {
      a0 += skt[k0 * LC_CH + p] * svt[v0 * LC_CH + p];
      a1 += skt[k1 * LC_CH + p] * svt[v1 * LC_CH + p];
    }
  }
  atomicAdd(&lamc[(b * KD + k0) * VD + v0], a0);
  atomicAdd(&lamc[(b * KD + k1) * VD + v1], a1);
}

// ---------------- fused finish: y = q_bn @ (lam_c + lam_p) ----------------
// block (48,4): one voxel per thread; grid (48*12, b)
__global__ void final_kernel(const float* __restrict__ proj,
                             const float* __restrict__ stats,
                             const float* __restrict__ lamc,
                             const float* __restrict__ Wpos,
                             const float* __restrict__ gq, const float* __restrict__ bq,
                             const float* __restrict__ gv, const float* __restrict__ bv,
                             float* __restrict__ out) {
  __shared__ float s_lamc[KD * VD];
  __shared__ float s_wpos[KD * 27];
  __shared__ float s_qsc[64], s_qsh[64], s_vsc[VD], s_vsh[VD];
  const int b = blockIdx.y;
  const int tid = threadIdx.y * 48 + threadIdx.x;   // 0..191

  for (int e = tid; e < KD * VD; e += 192) s_lamc[e] = lamc[b * KD * VD + e];
  for (int e = tid; e < KD * 27; e += 192) s_wpos[e] = Wpos[e];
  if (tid < 64) {
    float m = stats[2 * tid], va = stats[2 * tid + 1];
    float sc = gq[tid] * rsqrtf(va + EPS_BN);
    s_qsc[tid] = sc; s_qsh[tid] = bq[tid] - m * sc;
  } else if (tid < 96) {
    int v = tid - 64;
    float m = stats[2 * (64 + v)], va = stats[2 * (64 + v) + 1];
    float sc = gv[v] * rsqrtf(va + EPS_BN);
    s_vsc[v] = sc; s_vsh[v] = bv[v] - m * sc;
  }
  __syncthreads();

  const int d  = blockIdx.x / 12;
  const int hr = (blockIdx.x % 12) * 4 + threadIdx.y;
  const int w  = threadIdx.x;
  const int pos = (d * DDIM + hr) * DDIM + w;

  const float* pb = proj + (size_t)b * OTOT * NPOS;
  float qbn[64];
  #pragma unroll
  for (int o = 0; o < 64; ++o)
    qbn[o] = pb[(size_t)o * NPOS + pos] * s_qsc[o] + s_qsh[o];

  const float* vbase = pb + (size_t)V_OFF * NPOS;
  for (int v = 0; v < VD; ++v) {
    float lk[KD];
    #pragma unroll
    for (int k = 0; k < KD; ++k) lk[k] = s_lamc[k * VD + v];   // content lambda

    const float* vch = vbase + (size_t)v * NPOS;
    const float sc = s_vsc[v], sh = s_vsh[v];
    int tap = 0;
    for (int dz = -1; dz <= 1; ++dz) {
      int zz = d + dz; bool zok = (unsigned)zz < DDIM;
      for (int dy = -1; dy <= 1; ++dy) {
        int yy = hr + dy; bool yok = (unsigned)yy < DDIM;
        #pragma unroll
        for (int dx = -1; dx <= 1; ++dx, ++tap) {
          int xx = w + dx;
          float vn = 0.f;
          if (zok && yok && (unsigned)xx < DDIM)
            vn = vch[(zz * DDIM + yy) * DDIM + xx] * sc + sh;  // zero-pad post-BN
          #pragma unroll
          for (int k = 0; k < KD; ++k) lk[k] += s_wpos[k * 27 + tap] * vn;
        }
      }
    }
    #pragma unroll
    for (int hd = 0; hd < HEADS; ++hd) {
      float s = 0.f;
      #pragma unroll
      for (int k = 0; k < KD; ++k) s += qbn[hd * KD + k] * lk[k];
      out[((size_t)b * 128 + hd * VD + v) * NPOS + pos] = s;
    }
  }
}

// ---------------- host launcher ----------------
extern "C" void kernel_launch(void* const* d_in, const int* in_sizes, int n_in,
                              void* d_out, int out_size, void* d_ws, size_t ws_size,
                              hipStream_t stream) {
  const float* x    = (const float*)d_in[0];
  const float* Wq   = (const float*)d_in[1];
  const float* Wk   = (const float*)d_in[2];
  const float* Wv   = (const float*)d_in[3];
  const float* Wpos = (const float*)d_in[4];
  const float* gq   = (const float*)d_in[5];
  const float* bq   = (const float*)d_in[6];
  const float* gv   = (const float*)d_in[7];
  const float* bv   = (const float*)d_in[8];
  float* out = (float*)d_out;

  // workspace carve (all offsets 256B-aligned): ~156 MB total
  char* p = (char*)d_ws;
  unsigned short* xbf = (unsigned short*)p; p += (size_t)2 * C_IN * NPOS * sizeof(unsigned short); // 56.6 MB
  unsigned short* wbf = (unsigned short*)p; p += (size_t)OTOT * C_IN * sizeof(unsigned short);     // 28 KB
  float* proj   = (float*)p; p += (size_t)2 * OTOT * NPOS * sizeof(float);                         // 99.1 MB
  float* stats  = (float*)p; p += 96 * 2 * sizeof(float);
  float* sstats = (float*)p; p += 32 * 2 * sizeof(float);
  float* lamc   = (float*)p; p += 2 * KD * VD * sizeof(float);

  hipMemsetAsync(lamc, 0, 2 * KD * VD * sizeof(float), stream);

  convert_w_kernel<<<(OTOT * C_IN + 255) / 256, 256, 0, stream>>>(Wq, Wk, Wv, wbf);

  long long xtotal = (long long)2 * C_IN * NPOS;
  convert_x_kernel<<<(unsigned)((xtotal + 255) / 256), 256, 0, stream>>>(x, xbf, xtotal);

  proj_wmma_kernel<<<dim3(NPOS / 16, 2), 32, 0, stream>>>(xbf, wbf, proj);

  bn_stats_kernel<<<96, 256, 0, stream>>>(proj, stats);
  softmax_stats_kernel<<<32, 256, 0, stream>>>(proj, sstats);

  lam_c_kernel<<<dim3(NPOS / LC_POSB, 2), 256, 0, stream>>>(proj, stats, sstats, gv, bv, lamc);

  final_kernel<<<dim3(DDIM * (DDIM / 4), 2), dim3(48, 4, 1), 0, stream>>>(
      proj, stats, lamc, Wpos, gq, bq, gv, bv, out);
}
